// ScaledDotProductAttention_65592740545120
// MI455X (gfx1250) — matrix operand order repaired and checked by reading, not compiled, not run on hardware
//
#include <hip/hip_runtime.h>

typedef __attribute__((ext_vector_type(16))) _Float16 v16h;
typedef __attribute__((ext_vector_type(8)))  float    v8f;
typedef __attribute__((ext_vector_type(4)))  float    v4f;

#define SEQ   2048
#define HD    64
#define NBAT  4
#define NHEAD 16
#define BHTOT (NBAT*NHEAD)
#define QTILE 32
#define LSTRIDE 2065            /* 2048+17; 2065 % 64 == 17 -> conflict-free LDS patterns */
#define NEG_INF_F (-1e9f)
#define INV_TEMP  0.125f        /* 1/sqrt(64) */

// ---------------------------------------------------------------------------
// Pre-pass 1: K fp32 -> f16 (row-major, same layout), vectorized by 4.
// ---------------------------------------------------------------------------
__global__ void cvt_f32_to_f16_x4(const float* __restrict__ src,
                                  _Float16* __restrict__ dst, int n4) {
  int i = blockIdx.x * blockDim.x + threadIdx.x;
  if (i < n4) {
    v4f f = ((const v4f*)src)[i];
    dst[4*i+0] = (_Float16)f.x; dst[4*i+1] = (_Float16)f.y;
    dst[4*i+2] = (_Float16)f.z; dst[4*i+3] = (_Float16)f.w;
  }
}

// ---------------------------------------------------------------------------
// Pre-pass 2: V fp32 [bh][s][d] -> f16 transposed Vt [bh][d][s].
// One block = one (bh, 256-row s-tile). Coalesced reads, u32-packed writes.
// ---------------------------------------------------------------------------
__global__ __launch_bounds__(256)
void transpose_v_f16(const float* __restrict__ v, _Float16* __restrict__ Vt) {
  __shared__ float tile[256 * 65];            // [s_local][d], pad 65 -> conflict-free
  const int bh = blockIdx.y;
  const int s0 = blockIdx.x * 256;
  const int t  = threadIdx.x;
  const float* vsrc = v + (size_t)bh * SEQ * HD + (size_t)s0 * HD;
  for (int i = 0; i < 64; ++i) {              // 256*64 = 16384 elements, coalesced
    int lin = i * 256 + t;
    tile[(lin >> 6) * 65 + (lin & 63)] = vsrc[lin];
  }
  __syncthreads();
  const int dd2 = t >> 7;                     // 0/1
  const int tt  = t & 127;                    // packs s = 2*tt, 2*tt+1
  for (int d0 = 0; d0 < 64; d0 += 2) {
    int d = d0 + dd2;
    union { _Float16 h[2]; unsigned u; } p;
    p.h[0] = (_Float16)tile[(2*tt)     * 65 + d];
    p.h[1] = (_Float16)tile[(2*tt + 1) * 65 + d];
    ((unsigned*)(Vt + (size_t)(bh * HD + d) * SEQ + s0))[tt] = p.u;
  }
}

// ---------------------------------------------------------------------------
// Fused attention: logits (WMMA f16, fp32 accum) -> masked softmax (fp32)
// -> attn store (B128 non-temporal) -> out = (P_unnorm @ V) * rowinv (WMMA).
// One workgroup = one (b*h, 32-query tile). 256 threads = 8 waves.
// ---------------------------------------------------------------------------
__global__ __launch_bounds__(256)
void attn_fused_kernel(const float* __restrict__ q,
                       const int*   __restrict__ mask,
                       const _Float16* __restrict__ Kh,
                       const _Float16* __restrict__ Vt,
                       float* __restrict__ out,
                       float* __restrict__ attn) {
  extern __shared__ float lds[];              // [QTILE][LSTRIDE] logits + scratch
  float* red    = lds + QTILE * LSTRIDE;      // 256
  float* rowmax = red + 256;                  // 32
  float* rowinv = rowmax + QTILE;             // 32

  const int qtile = blockIdx.x;               // 0..63
  const int bh    = blockIdx.y;               // 0..63
  const int b     = bh >> 4;                  // bh / NHEAD
  const int qbase = qtile * QTILE;

  const int tid  = threadIdx.x;
  const int lane = tid & 31;
  const int wave = tid >> 5;
  const int hs   = (lane >> 4) & 1;           // half-wave select
  const int ln   = lane & 15;

  // ---------------- Phase A: raw logits -> LDS (fp32, unmasked) -----------
  {
    // Each wave owns key tiles j = wave + 8*i and computes BOTH 16-row blocks.
    // A layout: element e -> d = e + 16*hs.
    v16h qa00, qa01, qa10, qa11;
#pragma unroll
    for (int mbv = 0; mbv < 2; ++mbv) {
      const size_t qrow = (size_t)(bh * SEQ + qbase + mbv * 16 + ln) * HD;
#pragma unroll
      for (int e = 0; e < 16; ++e) {
        int d = e + 16 * hs;                  // 0..31
        _Float16 lo = (_Float16)(q[qrow + d]      * INV_TEMP);
        _Float16 hi = (_Float16)(q[qrow + d + 32] * INV_TEMP);
        if (mbv == 0) { qa00[e] = lo; qa01[e] = hi; }
        else          { qa10[e] = lo; qa11[e] = hi; }
      }
    }
    const v8f zero = {};
    for (int j = wave; j < SEQ / 16; j += 8) {
      // B layout: element e -> d = e + 16*hs -> contiguous 16 halves.
      const _Float16* krow = Kh + (size_t)(bh * SEQ + j * 16 + ln) * HD + 16 * hs;
      v16h b0 = *(const v16h*)(krow);
      v16h b1 = *(const v16h*)(krow + 32);
      // Unconditional: j+8 overshoot stays inside d_ws (Kh is followed by Vt),
      // keeps the hot loop branch-free (no exec-mask juggling).
      __builtin_prefetch(Kh + (size_t)(bh * SEQ + (j + 8) * 16 + ln) * HD, 0, 1);
      v8f c0 = __builtin_amdgcn_wmma_f32_16x16x32_f16(false, qa00, false, b0, (short)0, zero, false, false);
      v8f c1 = __builtin_amdgcn_wmma_f32_16x16x32_f16(false, qa10, false, b0, (short)0, zero, false, false);
      c0 = __builtin_amdgcn_wmma_f32_16x16x32_f16(false, qa01, false, b1, (short)0, c0, false, false);
      c1 = __builtin_amdgcn_wmma_f32_16x16x32_f16(false, qa11, false, b1, (short)0, c1, false, false);
      const int key = j * 16 + ln;
#pragma unroll
      for (int r = 0; r < 8; ++r) {           // C layout: row m = r + 8*hs
        int mrow = r + 8 * hs;
        lds[mrow * LSTRIDE + key]        = c0[r];
        lds[(16 + mrow) * LSTRIDE + key] = c1[r];
      }
    }
  }
  __syncthreads();

  // ----- Phase B1: apply mask (int4 coalesced) + row max; LDS <- masked ----
  const int m   = tid >> 3;                   // 0..31
  const int sub = tid & 7;
  {
    float* rowp = lds + m * LSTRIDE;
    const int4* mrow4 = (const int4*)(mask + (size_t)b * SEQ * SEQ +
                                      (size_t)(qbase + m) * SEQ);
    float lmax = -3.4e38f;
    for (int i = 0; i < SEQ / 32; ++i) {      // 64 iterations
      int k = 4 * sub + 32 * i;
      int4 mv = mrow4[sub + 8 * i];
      float x0 = mv.x ? rowp[k + 0] : NEG_INF_F;
      float x1 = mv.y ? rowp[k + 1] : NEG_INF_F;
      float x2 = mv.z ? rowp[k + 2] : NEG_INF_F;
      float x3 = mv.w ? rowp[k + 3] : NEG_INF_F;
      rowp[k + 0] = x0; rowp[k + 1] = x1; rowp[k + 2] = x2; rowp[k + 3] = x3;
      lmax = fmaxf(lmax, fmaxf(fmaxf(x0, x1), fmaxf(x2, x3)));
    }
    red[tid] = lmax;
    __syncthreads();
    if (tid < QTILE) {
      float mx = red[tid * 8];
#pragma unroll
      for (int s = 1; s < 8; ++s) mx = fmaxf(mx, red[tid * 8 + s]);
      rowmax[tid] = mx;
    }
    __syncthreads();
  }

  // ---------------- Phase B2: LDS <- exp(l - max); rowinv = 1/sum ---------
  {
    float* rowp = lds + m * LSTRIDE;
    const float mx = rowmax[m];
    float sum = 0.f;
    for (int i = 0; i < SEQ / 32; ++i) {
      int k = 4 * sub + 32 * i;
#pragma unroll
      for (int jj = 0; jj < 4; ++jj) {
        float e = __expf(rowp[k + jj] - mx);
        rowp[k + jj] = e;
        sum += e;
      }
    }
    red[tid] = sum;
    __syncthreads();
    if (tid < QTILE) {
      float s = 0.f;
#pragma unroll
      for (int k = 0; k < 8; ++k) s += red[tid * 8 + k];
      rowinv[tid] = 1.f / s;
    }
    __syncthreads();
  }

  // ------- Phase C: attn <- e * inv, B128 non-temporal stores --------------
  for (int mm = 0; mm < QTILE; ++mm) {
    const float inv = rowinv[mm];
    const float* rowp = lds + mm * LSTRIDE;
    float* arow = attn + (size_t)(bh * SEQ + qbase + mm) * SEQ;
#pragma unroll
    for (int i = 0; i < 2; ++i) {
      int k = 4 * tid + 1024 * i;
      v4f pv;
      pv.x = rowp[k + 0] * inv; pv.y = rowp[k + 1] * inv;
      pv.z = rowp[k + 2] * inv; pv.w = rowp[k + 3] * inv;
      __builtin_nontemporal_store(pv, (v4f*)(arow + k));
    }
  }
  __syncthreads();

  // ------ Phase D: out = (P_unnorm @ V) * rowinv (one 16x16 tile/wave) -----
  {
    const int mb = wave >> 2;                 // query-row block 0..1
    const int nb = wave & 3;                  // d-column block 0..3
    const float* prow = lds + (mb * 16 + ln) * LSTRIDE;
    v8f c = {};
    for (int kt = 0; kt < SEQ / 32; ++kt) {
      v16h av;
#pragma unroll
      for (int e = 0; e < 16; ++e) {          // A gather (stride 17 mod 64: conflict-free)
        int kk = kt * 32 + e + 16 * hs;
        av[e] = (_Float16)prow[kk];           // raw e values; row scale applied at the end
      }
      // B operand from transposed Vt: element e -> k = kt*32 + e + 16*hs, contiguous.
      const _Float16* vrow = Vt + (size_t)(bh * HD + nb * 16 + ln) * SEQ + kt * 32 + 16 * hs;
      v16h bv = *(const v16h*)vrow;
      c = __builtin_amdgcn_wmma_f32_16x16x32_f16(false, av, false, bv, (short)0, c, false, false);
    }
#pragma unroll
    for (int r = 0; r < 8; ++r) {
      int mrow = mb * 16 + r + 8 * hs;
      out[(size_t)(bh * SEQ + qbase + mrow) * HD + nb * 16 + ln] = c[r] * rowinv[mrow];
    }
  }
}

// ---------------------------------------------------------------------------
extern "C" void kernel_launch(void* const* d_in, const int* in_sizes, int n_in,
                              void* d_out, int out_size, void* d_ws, size_t ws_size,
                              hipStream_t stream) {
  const float* q    = (const float*)d_in[0];
  const float* k    = (const float*)d_in[1];
  const float* v    = (const float*)d_in[2];
  const int*   mask = (const int*)  d_in[3];

  float* out  = (float*)d_out;                       // [B,H,S,D]
  float* attn = out + (size_t)BHTOT * SEQ * HD;      // [B,H,S,S]

  const size_t nelem = (size_t)BHTOT * SEQ * HD;     // 8,388,608
  _Float16* Kh = (_Float16*)d_ws;                    // row-major f16 K
  _Float16* Vt = Kh + nelem;                         // transposed f16 V [bh][d][s]

  const int n4 = (int)(nelem / 4);
  cvt_f32_to_f16_x4<<<(n4 + 255) / 256, 256, 0, stream>>>(k, Kh, n4);
  transpose_v_f16<<<dim3(SEQ / 256, BHTOT), 256, 0, stream>>>(v, Vt);

  dim3 grid(SEQ / QTILE, BHTOT);                     // 64 x 64
  size_t shmem = (size_t)(QTILE * LSTRIDE + 256 + QTILE + QTILE) * sizeof(float);
  attn_fused_kernel<<<grid, 256, shmem, stream>>>(q, mask, Kh, Vt, out, attn);
}
